// CaptionDecoder_18459769438865
// MI455X (gfx1250) — matrix operand order repaired and checked by reading, not compiled, not run on hardware
//
#include <hip/hip_runtime.h>

typedef __attribute__((ext_vector_type(16))) _Float16 v16h;
typedef __attribute__((ext_vector_type(8)))  float    v8f;

__device__ __forceinline__ float sigf(float x) { return 1.0f / (1.0f + expf(-x)); }

// ------------------------- preprocessing kernels -------------------------
__global__ void k_cvt_f16(const float* __restrict__ s, _Float16* __restrict__ d, int n) {
  int i = blockIdx.x * blockDim.x + threadIdx.x;
  if (i < n) d[i] = (_Float16)s[i];
}

__global__ void k_pad_wf(const float* __restrict__ wf, _Float16* __restrict__ d) {
  int i = blockIdx.x * blockDim.x + threadIdx.x;  // 10048*512
  if (i >= 10048 * 512) return;
  int r = i >> 9, c = i & 511;
  d[i] = (r < 10000) ? (_Float16)wf[r * 512 + c] : (_Float16)0.0f;
}

__global__ void k_wcat(const float* __restrict__ wih, const float* __restrict__ whh,
                       _Float16* __restrict__ d) {
  int i = blockIdx.x * blockDim.x + threadIdx.x;  // 2048*3072
  if (i >= 2048 * 3072) return;
  int r = i / 3072, c = i % 3072;
  float v = (c < 2560) ? wih[(long)r * 2560 + c] : whh[(long)r * 512 + (c - 2560)];
  d[i] = (_Float16)v;
}

__global__ void k_bsum(const float* __restrict__ a, const float* __restrict__ b,
                       float* __restrict__ d) {
  int i = blockIdx.x * blockDim.x + threadIdx.x;
  if (i < 2048) d[i] = a[i] + b[i];
}

__global__ void k_embed(const int* __restrict__ cap, const float* __restrict__ tab,
                        _Float16* __restrict__ d) {
  int i = blockIdx.x * blockDim.x + threadIdx.x;  // 64*32*512
  if (i >= 64 * 32 * 512) return;
  int bt = i >> 9, e = i & 511;
  d[i] = (_Float16)tab[(long)cap[bt] * 512 + e];
}

__global__ void k_init_state(float* h, float* c, _Float16* h16) {
  int i = blockIdx.x * blockDim.x + threadIdx.x;
  if (i < 64 * 512) { h[i] = 0.f; c[i] = 0.f; h16[i] = (_Float16)0.f; }
}

__global__ void k_final(const float* __restrict__ h, const float* __restrict__ c,
                        float* __restrict__ oh, float* __restrict__ oc) {
  int i = blockIdx.x * blockDim.x + threadIdx.x;
  if (i < 64 * 512) { oh[i] = h[i]; oc[i] = c[i]; }
}

// ------------------------- WMMA GEMM: C[M,N] = A[M,K] * W[N,K]^T + bias -------------------------
// Workgroup: 4 waves -> 64(M) x 64(N) macro tile; wave -> 16x64 (4 accumulators).
// A layout per lane (16x32 f16):  m = lane&15, half = lane>>4,
//   elems 0..7  <- K = kb + 8*half + e        (16B contiguous)
//   elems 8..15 <- K = kb + 16 + 8*half + e   (16B contiguous)
// B layout per lane (32x16 f16):  n-col = lane&15,
//   elems 0..7  <- K = kb + 8*half + e        (16B contiguous from weight row n)
//   elems 8..15 <- K = kb + 16 + 8*half + e   (16B contiguous from weight row n)
template <bool A_IS_F32>
__global__ void gemm_wmma_f16(const void* __restrict__ Ap, const _Float16* __restrict__ W,
                              float* __restrict__ C, const float* __restrict__ bias,
                              int K, long ldc, long cbase, int nlimit) {
  const int lane  = threadIdx.x & 31;
  const int wave  = threadIdx.x >> 5;
  const int mbase = blockIdx.x * 64 + wave * 16;
  const int nbase = blockIdx.y * 64;
  const int mrow  = mbase + (lane & 15);
  const int hf    = lane >> 4;

  v8f acc[4] = {};
  const float*    Af = (const float*)Ap;
  const _Float16* Ah = (const _Float16*)Ap;

  for (int kb = 0; kb < K; kb += 32) {
    v16h a;
    if (A_IS_F32) {
      const float* ar = Af + (long)mrow * K + kb + hf * 8;
#pragma unroll
      for (int e = 0; e < 8; ++e) a[e] = (_Float16)ar[e];
#pragma unroll
      for (int e = 0; e < 8; ++e) a[8 + e] = (_Float16)ar[16 + e];
    } else {
      const _Float16* ar = Ah + (long)mrow * K + kb + hf * 8;
#pragma unroll
      for (int e = 0; e < 8; ++e) a[e] = ar[e];
#pragma unroll
      for (int e = 0; e < 8; ++e) a[8 + e] = ar[16 + e];
    }
#pragma unroll
    for (int s = 0; s < 4; ++s) {
      const _Float16* br = W + (long)(nbase + s * 16 + (lane & 15)) * K + kb + hf * 8;
      // Speculative near-cache prefetch of next K-block of this weight row.
      // Unconditional (drop-on-fault; worst case lands in adjacent ws region).
      __builtin_prefetch(br + 32, 0, 3);
      v16h b;
#pragma unroll
      for (int e = 0; e < 8; ++e) b[e] = br[e];
#pragma unroll
      for (int e = 0; e < 8; ++e) b[8 + e] = br[16 + e];
      acc[s] = __builtin_amdgcn_wmma_f32_16x16x32_f16(
          /*neg_a=*/false, a, /*neg_b=*/false, b,
          /*c_mod=*/(short)0, acc[s], /*reuse_a=*/false, /*reuse_b=*/false);
    }
  }
  // C layout: v8f elem r -> row mbase + r + 8*half, col nbase + s*16 + (lane&15)
#pragma unroll
  for (int s = 0; s < 4; ++s) {
    int n = nbase + s * 16 + (lane & 15);
    if (n < nlimit) {
      float bv = bias ? bias[n] : 0.0f;
#pragma unroll
      for (int r = 0; r < 8; ++r) {
        long mm = mbase + r + hf * 8;
        C[cbase + mm * ldc + n] = acc[s][r] + bv;
      }
    }
  }
}

// ------------------------- fused attention: energy + softmax + context -------------------------
__global__ void k_attention(const float* __restrict__ enc_map, const float* __restrict__ dec_map,
                            const float* __restrict__ enc_out, const float* __restrict__ Wa,
                            const float* __restrict__ ba, const _Float16* __restrict__ embs,
                            const _Float16* __restrict__ h16, float* __restrict__ attn_out,
                            _Float16* __restrict__ xcat, int t) {
  __shared__ float s_dec[512], s_wa[512], s_e[196], s_al[196], s_inv;
  const int b = blockIdx.x, tid = threadIdx.x;
  const int wid = tid >> 5, lane = tid & 31;
  for (int i = tid; i < 512; i += 256) { s_dec[i] = dec_map[b * 512 + i]; s_wa[i] = Wa[i]; }
  __syncthreads();
  // e[p] = sum_a tanh(enc_map + dec_map) * Wa  (one wave per p)
  for (int p = wid; p < 196; p += 8) {
    const float* row = enc_map + ((long)b * 196 + p) * 512;
    float sum = 0.f;
    for (int a2 = lane; a2 < 512; a2 += 32) sum += tanhf(row[a2] + s_dec[a2]) * s_wa[a2];
#pragma unroll
    for (int off = 16; off > 0; off >>= 1) sum += __shfl_xor(sum, off, 32);
    if (lane == 0) s_e[p] = sum + ba[0];
  }
  __syncthreads();
  if (wid == 0) {  // softmax stats on one wave
    float mx = -1e30f;
    for (int p = lane; p < 196; p += 32) mx = fmaxf(mx, s_e[p]);
#pragma unroll
    for (int off = 16; off > 0; off >>= 1) mx = fmaxf(mx, __shfl_xor(mx, off, 32));
    float sm = 0.f;
    for (int p = lane; p < 196; p += 32) { float v = expf(s_e[p] - mx); s_al[p] = v; sm += v; }
#pragma unroll
    for (int off = 16; off > 0; off >>= 1) sm += __shfl_xor(sm, off, 32);
    if (lane == 0) s_inv = 1.0f / sm;
  }
  __syncthreads();
  float inv = s_inv;
  for (int p = tid; p < 196; p += 256) {
    float al = s_al[p] * inv;
    s_al[p] = al;
    attn_out[((long)b * 32 + t) * 196 + p] = al;
  }
  __syncthreads();
  // context -> xcat[:,0:2048] (f16); then append emb and h
  for (int c = tid; c < 2048; c += 256) {
    float acc = 0.f;
    const float* col = enc_out + (long)b * 196 * 2048 + c;
    for (int p = 0; p < 196; ++p) acc += s_al[p] * col[(long)p * 2048];
    xcat[(long)b * 3072 + c] = (_Float16)acc;
  }
  for (int j = tid; j < 512; j += 256) {
    xcat[(long)b * 3072 + 2048 + j] = embs[((long)b * 32 + t) * 512 + j];
    xcat[(long)b * 3072 + 2560 + j] = h16[b * 512 + j];
  }
}

// ------------------------- LSTM pointwise (gate order i,f,g,o) -------------------------
__global__ void k_lstm(const float* __restrict__ gates, float* __restrict__ h,
                       float* __restrict__ c, _Float16* __restrict__ h16) {
  int i = blockIdx.x * blockDim.x + threadIdx.x;
  if (i >= 64 * 512) return;
  int b = i >> 9, j = i & 511;
  const float* g = gates + (long)b * 2048;
  float ig = sigf(g[j]), fg = sigf(g[512 + j]);
  float gg = tanhf(g[1024 + j]), og = sigf(g[1536 + j]);
  float cn = fg * c[i] + ig * gg;
  float hn = og * tanhf(cn);
  c[i] = cn; h[i] = hn; h16[i] = (_Float16)hn;
}

// ------------------------- host driver -------------------------
extern "C" void kernel_launch(void* const* d_in, const int* in_sizes, int n_in,
                              void* d_out, int out_size, void* d_ws, size_t ws_size,
                              hipStream_t stream) {
  (void)in_sizes; (void)n_in; (void)out_size; (void)ws_size;
  const float* enc_out  = (const float*)d_in[0];
  const int*   captions = (const int*)d_in[1];
  const float* emb_tab  = (const float*)d_in[2];
  const float* We  = (const float*)d_in[3];
  const float* be  = (const float*)d_in[4];
  const float* Wd  = (const float*)d_in[5];
  const float* bd  = (const float*)d_in[6];
  const float* Wa  = (const float*)d_in[7];
  const float* ba  = (const float*)d_in[8];
  const float* Wih = (const float*)d_in[9];
  const float* bih = (const float*)d_in[10];
  const float* Whh = (const float*)d_in[11];
  const float* bhh = (const float*)d_in[12];
  const float* Wf  = (const float*)d_in[13];
  const float* bf  = (const float*)d_in[14];

  char* ws = (char*)d_ws;
  size_t off = 0;
  auto take = [&](size_t bytes) -> char* {
    char* p = ws + off;
    off = (off + bytes + 255) & ~(size_t)255;
    return p;
  };
  float*    enc_map = (float*)take(12544ull * 512 * 4);    // [B*P, ATTN]
  _Float16* We16    = (_Float16*)take(512ull * 2048 * 2);
  _Float16* Wd16    = (_Float16*)take(512ull * 512 * 2);
  _Float16* Wcat    = (_Float16*)take(2048ull * 3072 * 2); // [W_ih | W_hh]
  _Float16* Wf16    = (_Float16*)take(10048ull * 512 * 2); // padded rows
  float*    bsum    = (float*)take(2048ull * 4);
  _Float16* embs    = (_Float16*)take(64ull * 32 * 512 * 2);
  _Float16* xcat    = (_Float16*)take(64ull * 3072 * 2);   // [ctx | emb | h]
  float*    h       = (float*)take(64ull * 512 * 4);
  float*    c       = (float*)take(64ull * 512 * 4);
  _Float16* h16     = (_Float16*)take(64ull * 512 * 2);
  float*    dec_map = (float*)take(64ull * 512 * 4);
  float*    gates   = (float*)take(64ull * 2048 * 4);

  float* out        = (float*)d_out;
  float* out_logits = out;                       // [B,T,V]
  float* out_h      = out + 64ull * 32 * 10000;  // [1,B,H]
  float* out_c      = out_h + 64 * 512;          // [1,B,H]
  float* out_attn   = out_c + 64 * 512;          // [B,T,P]

  const int TB = 256;
  k_cvt_f16<<<(512 * 2048 + TB - 1) / TB, TB, 0, stream>>>(We, We16, 512 * 2048);
  k_cvt_f16<<<(512 * 512 + TB - 1) / TB, TB, 0, stream>>>(Wd, Wd16, 512 * 512);
  k_pad_wf<<<(10048 * 512 + TB - 1) / TB, TB, 0, stream>>>(Wf, Wf16);
  k_wcat<<<(2048 * 3072 + TB - 1) / TB, TB, 0, stream>>>(Wih, Whh, Wcat);
  k_bsum<<<(2048 + TB - 1) / TB, TB, 0, stream>>>(bih, bhh, bsum);
  k_embed<<<(64 * 32 * 512 + TB - 1) / TB, TB, 0, stream>>>(captions, emb_tab, embs);
  k_init_state<<<(64 * 512 + TB - 1) / TB, TB, 0, stream>>>(h, c, h16);

  // enc_map = encoder_out @ We^T + be   (A is f32, converted on the fly)
  gemm_wmma_f16<true><<<dim3(196, 8), 128, 0, stream>>>(
      enc_out, We16, enc_map, be, 2048, 512, 0, 512);

  for (int t = 0; t < 32; ++t) {
    // dec_map = h @ Wd^T + bd
    gemm_wmma_f16<false><<<dim3(1, 8), 128, 0, stream>>>(
        h16, Wd16, dec_map, bd, 512, 512, 0, 512);
    // attention: energy + softmax + context; builds xcat = [ctx, emb_t, h]
    k_attention<<<64, 256, 0, stream>>>(enc_map, dec_map, enc_out, Wa, ba,
                                        embs, h16, out_attn, xcat, t);
    // gates = xcat @ [W_ih|W_hh]^T + (b_ih + b_hh)
    gemm_wmma_f16<false><<<dim3(1, 32), 128, 0, stream>>>(
        xcat, Wcat, gates, bsum, 3072, 2048, 0, 2048);
    // LSTM pointwise -> h, c, h16
    k_lstm<<<(64 * 512 + TB - 1) / TB, TB, 0, stream>>>(gates, h, c, h16);
    // logits = h_new @ Wf^T + bf, written straight into outputs[:, t, :]
    gemm_wmma_f16<false><<<dim3(1, 157), 128, 0, stream>>>(
        h16, Wf16, out_logits, bf, 512, 32l * 10000, (long)t * 10000, 10000);
  }
  k_final<<<(64 * 512 + TB - 1) / TB, TB, 0, stream>>>(h, c, out_h, out_c);
}
